// MultiHeadAttention_10015863734738
// MI455X (gfx1250) — matrix operand order repaired and checked by reading, not compiled, not run on hardware
//
#include <hip/hip_runtime.h>

// ---------------------------------------------------------------------------
// MHA for B=4, T=2048, D=1024, N=16 heads (HD=64) on gfx1250 (MI455X).
// All matmuls via v_wmma_f32_16x16x32_bf16 (wave32).
// ---------------------------------------------------------------------------

typedef __attribute__((ext_vector_type(16))) __bf16 v16bf;
typedef __attribute__((ext_vector_type(8)))  float  v8f;

#define D_MODEL 1024
#define T_SEQ   2048
#define B_BATCH 4
#define N_HEADS 16
#define HD      64
#define M_ROWS  (B_BATCH * T_SEQ)   // 8192

union FragAB { v16bf v; unsigned int u[8]; };

// A-matrix (16xK) fragment contraction index for vgpr-pair j, lane half
__device__ __forceinline__ int ka0(int j, int half) {
  return 2 * (j & 3) + 8 * half + 16 * (j >> 2);
}
// Contiguous contraction index (k = 2j + 16*half); unused, B fragments use ka0
__device__ __forceinline__ int kb0(int j, int half) {
  return 2 * j + 16 * half;
}

__device__ __forceinline__ unsigned pack_bf2(float x, float y) {
  union { __bf16 h[2]; unsigned u; } p;
  p.h[0] = (__bf16)x;
  p.h[1] = (__bf16)y;
  return p.u;
}

__device__ __forceinline__ v8f vzero8() {
  v8f z = {0.f, 0.f, 0.f, 0.f, 0.f, 0.f, 0.f, 0.f};
  return z;
}

// ---------------------------------------------------------------------------
// GEMM + bias: C[M=8192][1024] = A[8192][1024] * W[1024][1024] + bias
// MODE 0: store bf16 row-major [M][D]
// MODE 1: store bf16 per-head transposed [B][N][HD][T]   (for V)
// MODE 2: store f32 row-major [M][D]                      (final output)
// Block: 256 thr (8 waves), tile 128(M) x 128(N), K-step 32.
// Wave (wm=wid&3 -> 32 rows, wn=wid>>2 -> 64 cols): 2x4 = 8 WMMA tiles.
// ---------------------------------------------------------------------------
template <int MODE>
__global__ __launch_bounds__(256) void gemm_bias_wmma(
    const float* __restrict__ A, const float* __restrict__ W,
    const float* __restrict__ bias, void* __restrict__ out)
{
  __shared__ unsigned ldsA[128][17];   // 128 rows x 32 bf16 (16 dwords) + pad
  __shared__ unsigned ldsB[128][17];   // 128 cols x 32 bf16 (k-pairs) + pad

  const int tid  = threadIdx.x;
  const int row0 = blockIdx.y * 128;
  const int col0 = blockIdx.x * 128;
  const int wid  = tid >> 5;
  const int lane = tid & 31;
  const int wmR  = (wid & 3) * 32;     // wave row base within block tile
  const int wnC  = (wid >> 2) * 64;    // wave col base within block tile
  const int half = lane >> 4;
  const int nl   = lane & 15;

  v8f acc[2][4];
#pragma unroll
  for (int mt = 0; mt < 2; ++mt)
#pragma unroll
    for (int nt = 0; nt < 4; ++nt) acc[mt][nt] = vzero8();

  // staging maps (computed once)
  const int ar = tid >> 1, aq = tid & 1;          // A: row, 16-col group
  const int cp = tid >> 4, g  = tid & 15;         // W: k-pair, 8-col group

  for (int kb = 0; kb < D_MODEL; kb += 32) {
    __syncthreads();
    // ---- stage A tile (128 x 32 fp32 -> bf16) ----
    {
      const float* base = A + (size_t)(row0 + ar) * D_MODEL + kb + aq * 16;
      const float4* src = (const float4*)base;
      float4 a0 = src[0], a1 = src[1], a2 = src[2], a3 = src[3];
      const int c0 = aq * 8;
      ldsA[ar][c0 + 0] = pack_bf2(a0.x, a0.y);
      ldsA[ar][c0 + 1] = pack_bf2(a0.z, a0.w);
      ldsA[ar][c0 + 2] = pack_bf2(a1.x, a1.y);
      ldsA[ar][c0 + 3] = pack_bf2(a1.z, a1.w);
      ldsA[ar][c0 + 4] = pack_bf2(a2.x, a2.y);
      ldsA[ar][c0 + 5] = pack_bf2(a2.z, a2.w);
      ldsA[ar][c0 + 6] = pack_bf2(a3.x, a3.y);
      ldsA[ar][c0 + 7] = pack_bf2(a3.z, a3.w);
      if (kb + 32 < D_MODEL)
        __builtin_prefetch(base + 32, 0, 1);      // global_prefetch next tile
    }
    // ---- stage W tile transposed (32 x 128 fp32 -> bf16 pairs along K) ----
    {
      const float* b0 = W + (size_t)(kb + 2 * cp) * D_MODEL + col0 + g * 8;
      const float4* s0 = (const float4*)b0;
      const float4* s1 = (const float4*)(b0 + D_MODEL);
      float4 x0 = s0[0], x1 = s0[1], y0 = s1[0], y1 = s1[1];
      const int nb = g * 8;
      ldsB[nb + 0][cp] = pack_bf2(x0.x, y0.x);
      ldsB[nb + 1][cp] = pack_bf2(x0.y, y0.y);
      ldsB[nb + 2][cp] = pack_bf2(x0.z, y0.z);
      ldsB[nb + 3][cp] = pack_bf2(x0.w, y0.w);
      ldsB[nb + 4][cp] = pack_bf2(x1.x, y1.x);
      ldsB[nb + 5][cp] = pack_bf2(x1.y, y1.y);
      ldsB[nb + 6][cp] = pack_bf2(x1.z, y1.z);
      ldsB[nb + 7][cp] = pack_bf2(x1.w, y1.w);
      if (kb + 32 < D_MODEL)
        __builtin_prefetch(b0 + 32 * D_MODEL, 0, 1);
    }
    __syncthreads();

    // ---- fragments + 8 WMMAs ----
    FragAB afrag[2];
#pragma unroll
    for (int mt = 0; mt < 2; ++mt)
#pragma unroll
      for (int j = 0; j < 8; ++j)
        afrag[mt].u[j] = ldsA[wmR + mt * 16 + nl][ka0(j, half) >> 1];

#pragma unroll
    for (int nt = 0; nt < 4; ++nt) {
      FragAB bfrag;
#pragma unroll
      for (int j = 0; j < 8; ++j)
        bfrag.u[j] = ldsB[wnC + nt * 16 + nl][ka0(j, half) >> 1];   // ka0>>1
#pragma unroll
      for (int mt = 0; mt < 2; ++mt)
        acc[mt][nt] = __builtin_amdgcn_wmma_f32_16x16x32_bf16(
            false, afrag[mt].v, false, bfrag.v, (short)0, acc[mt][nt],
            false, false);
    }
  }

  // ---- epilogue: bias + store (straight-line, MODE is compile-time) ----
#pragma unroll
  for (int nt = 0; nt < 4; ++nt) {
    const int col = col0 + wnC + nt * 16 + nl;
    const float bv = bias[col];
#pragma unroll
    for (int mt = 0; mt < 2; ++mt) {
#pragma unroll
      for (int r = 0; r < 8; ++r) {
        const int row = row0 + wmR + mt * 16 + r + 8 * half;
        const float v = acc[mt][nt][r] + bv;
        if (MODE == 0) {
          ((__bf16*)out)[(size_t)row * D_MODEL + col] = (__bf16)v;
        } else if (MODE == 1) {
          const int b  = row >> 11;       // /T
          const int t  = row & 2047;
          const int nh = col >> 6;        // /HD
          const int hd = col & 63;
          ((__bf16*)out)[(((size_t)(b * N_HEADS + nh) * HD + hd) << 11) + t] =
              (__bf16)v;
        } else {
          ((float*)out)[(size_t)row * D_MODEL + col] = v;
        }
      }
    }
  }
}

// ---------------------------------------------------------------------------
// Causal flash attention. One wave per (b, head, 16-row q tile).
// Q: bf16 [B*T][D], K: bf16 [B*T][D], Vt: bf16 [B][N][HD][T], O: f32 [B*T][D].
// 1024 blocks x 256 thr (8 independent waves/block, no block barriers).
// ---------------------------------------------------------------------------
__global__ __launch_bounds__(256) void attn_wmma(
    const __bf16* __restrict__ Q, const __bf16* __restrict__ K,
    const __bf16* __restrict__ Vt, float* __restrict__ Oout)
{
  __shared__ __bf16 pbuf[8][16][32];   // per-wave P transpose scratch (8 KB)

  const int tid  = threadIdx.x;
  const int wid  = tid >> 5;
  const int lane = tid & 31;
  const int half = lane >> 4;
  const int nl   = lane & 15;

  const int task  = blockIdx.x * 8 + wid;   // 0..8191
  const int b     = task >> 11;             // / (N_HEADS * T/16)
  const int rem   = task & 2047;
  const int nh    = rem >> 7;               // / (T/16)
  const int qt    = rem & 127;
  const int qbase = qt * 16;

  // Q A-fragments (16 rows x 64 hd), resident for the whole tile
  FragAB qf[2];
  {
    const __bf16* qptr =
        Q + (size_t)(b * T_SEQ + qbase + nl) * D_MODEL + nh * HD;
#pragma unroll
    for (int cc = 0; cc < 2; ++cc)
#pragma unroll
      for (int j = 0; j < 8; ++j)
        qf[cc].u[j] = *(const unsigned*)(qptr + 32 * cc + ka0(j, half));
  }

  v8f o[4];
#pragma unroll
  for (int nt = 0; nt < 4; ++nt) o[nt] = vzero8();
  float mrow[8], lrow[8];
#pragma unroll
  for (int r = 0; r < 8; ++r) { mrow[r] = -1e30f; lrow[r] = 0.f; }

  const float inv_scale = 0.03125f;   // 1/sqrt(D) = 1/32
  const int kend = qbase + 15;        // last needed key (causal)

  for (int kb = 0; kb <= kend; kb += 32) {
    // ---- S = Q * K^T for 32 keys (two 16-key subtiles) ----
    v8f s0 = vzero8(), s1 = vzero8();
    {
      const __bf16* kp0 =
          K + (size_t)(b * T_SEQ + kb + nl) * D_MODEL + nh * HD;
      const __bf16* kp1 =
          K + (size_t)(b * T_SEQ + kb + 16 + nl) * D_MODEL + nh * HD;
      FragAB k0a, k0b, k1a, k1b;
#pragma unroll
      for (int j = 0; j < 8; ++j) {
        const int h0 = ka0(j, half);
        k0a.u[j] = *(const unsigned*)(kp0 + h0);
        k0b.u[j] = *(const unsigned*)(kp0 + 32 + h0);
        k1a.u[j] = *(const unsigned*)(kp1 + h0);
        k1b.u[j] = *(const unsigned*)(kp1 + 32 + h0);
      }
      s0 = __builtin_amdgcn_wmma_f32_16x16x32_bf16(
          false, qf[0].v, false, k0a.v, (short)0, s0, false, false);
      s0 = __builtin_amdgcn_wmma_f32_16x16x32_bf16(
          false, qf[1].v, false, k0b.v, (short)0, s0, false, false);
      s1 = __builtin_amdgcn_wmma_f32_16x16x32_bf16(
          false, qf[0].v, false, k1a.v, (short)0, s1, false, false);
      s1 = __builtin_amdgcn_wmma_f32_16x16x32_bf16(
          false, qf[1].v, false, k1b.v, (short)0, s1, false, false);
    }

    // ---- online softmax over the 32-key chunk ----
    const int key0 = kb + nl;
    const int key1 = kb + 16 + nl;
#pragma unroll
    for (int r = 0; r < 8; ++r) {
      const int qrow = qbase + r + 8 * half;
      float l0 = (key0 <= qrow) ? s0[r] * inv_scale : -1e30f;
      float l1 = (key1 <= qrow) ? s1[r] * inv_scale : -1e30f;
      float mx = fmaxf(l0, l1);
      mx = fmaxf(mx, __shfl_xor(mx, 1, 32));
      mx = fmaxf(mx, __shfl_xor(mx, 2, 32));
      mx = fmaxf(mx, __shfl_xor(mx, 4, 32));
      mx = fmaxf(mx, __shfl_xor(mx, 8, 32));
      const float mnew = fmaxf(mrow[r], mx);
      const float corr = __expf(mrow[r] - mnew);
      const float p0 = __expf(l0 - mnew);
      const float p1 = __expf(l1 - mnew);
      float ps = p0 + p1;
      ps += __shfl_xor(ps, 1, 32);
      ps += __shfl_xor(ps, 2, 32);
      ps += __shfl_xor(ps, 4, 32);
      ps += __shfl_xor(ps, 8, 32);
      lrow[r] = lrow[r] * corr + ps;
      mrow[r] = mnew;
#pragma unroll
      for (int nt = 0; nt < 4; ++nt) o[nt][r] *= corr;
      // C-layout -> LDS (row-major 16x32) for the A-fragment transpose
      pbuf[wid][r + 8 * half][nl]      = (__bf16)p0;
      pbuf[wid][r + 8 * half][16 + nl] = (__bf16)p1;
    }

    // ---- P as A-fragment (wave-private LDS, DS ops are in-order) ----
    FragAB pf;
#pragma unroll
    for (int j = 0; j < 8; ++j)
      pf.u[j] = *(const unsigned*)&pbuf[wid][nl][ka0(j, half)];

    // ---- O += P * V (V pairs contiguous along keys thanks to Vt layout) ----
    const __bf16* vbase = Vt + (size_t)(b * N_HEADS + nh) * HD * T_SEQ;
#pragma unroll
    for (int nt = 0; nt < 4; ++nt) {
      const int hd = nt * 16 + nl;
      const __bf16* vptr = vbase + (size_t)hd * T_SEQ + kb;
      FragAB vf;
#pragma unroll
      for (int j = 0; j < 8; ++j)
        vf.u[j] = *(const unsigned*)(vptr + ka0(j, half));
      o[nt] = __builtin_amdgcn_wmma_f32_16x16x32_bf16(
          false, pf.v, false, vf.v, (short)0, o[nt], false, false);
    }
  }

  // ---- normalize and store fp32 [B*T][D] ----
#pragma unroll
  for (int r = 0; r < 8; ++r) {
    const float inv = 1.0f / lrow[r];
    const int row = b * T_SEQ + qbase + r + 8 * half;
#pragma unroll
    for (int nt = 0; nt < 4; ++nt)
      Oout[(size_t)row * D_MODEL + nh * HD + nt * 16 + nl] = o[nt][r] * inv;
  }
}

// ---------------------------------------------------------------------------
// Launch
// ---------------------------------------------------------------------------
extern "C" void kernel_launch(void* const* d_in, const int* in_sizes, int n_in,
                              void* d_out, int out_size, void* d_ws,
                              size_t ws_size, hipStream_t stream)
{
  (void)in_sizes; (void)n_in; (void)out_size; (void)ws_size;

  const float* query = (const float*)d_in[0];
  const float* key   = (const float*)d_in[1];
  const float* value = (const float*)d_in[2];
  const float* wq_w  = (const float*)d_in[3];
  const float* wq_b  = (const float*)d_in[4];
  const float* wk_w  = (const float*)d_in[5];
  const float* wk_b  = (const float*)d_in[6];
  const float* wv_w  = (const float*)d_in[7];
  const float* wv_b  = (const float*)d_in[8];
  const float* wo_w  = (const float*)d_in[9];
  const float* wo_b  = (const float*)d_in[10];

  char* ws = (char*)d_ws;
  __bf16* Qbf = (__bf16*)(ws + (size_t)0);                   // 16 MB
  __bf16* Kbf = (__bf16*)(ws + (size_t)16 * 1024 * 1024);    // 16 MB
  __bf16* Vt  = (__bf16*)(ws + (size_t)32 * 1024 * 1024);    // 16 MB
  float*  Oat = (float* )(ws + (size_t)48 * 1024 * 1024);    // 32 MB

  dim3 grid_g(8, 64);      // N/128 x M/128
  dim3 blk(256);

  gemm_bias_wmma<0><<<grid_g, blk, 0, stream>>>(query, wq_w, wq_b, Qbf);
  gemm_bias_wmma<0><<<grid_g, blk, 0, stream>>>(key,   wk_w, wk_b, Kbf);
  gemm_bias_wmma<1><<<grid_g, blk, 0, stream>>>(value, wv_w, wv_b, Vt);

  attn_wmma<<<dim3(1024), blk, 0, stream>>>(Qbf, Kbf, Vt, Oat);

  gemm_bias_wmma<2><<<grid_g, blk, 0, stream>>>(Oat, wo_w, wo_b, d_out);
}
